// YiJingAttention_15221364097194
// MI455X (gfx1250) — compile-verified, matrix operand order checked by reading, not run on hardware
//
#include <hip/hip_runtime.h>
#include <hip/hip_bf16.h>

typedef __attribute__((ext_vector_type(16))) _Float16 v16h;
typedef __attribute__((ext_vector_type(8)))  float    v8f;
typedef __attribute__((ext_vector_type(4)))  int      v4i;

union FragH { v16h h; uint32_t u[8]; };
union PackH { uint32_t u; _Float16 h[2]; };

#define B_  4
#define T_  1024
#define D_  1024
#define H_  8
#define HD_ 128

// ---------------------------------------------------------------------------
// Async global->LDS copy (CDNA5 GLOBAL_LOAD_ASYNC_TO_LDS_B128, ASYNCcnt) with
// a register-mediated fallback when the builtin is unavailable.
// Probe result: builtin takes a typed b128 payload pointer (int4*).
// ---------------------------------------------------------------------------
#if __has_builtin(__builtin_amdgcn_global_load_async_to_lds_b128)
#define HAVE_ASYNC 1
__device__ __forceinline__ void async_cp16(const _Float16* g, _Float16* l) {
    __builtin_amdgcn_global_load_async_to_lds_b128(
        (__attribute__((address_space(1))) v4i*)g,
        (__attribute__((address_space(3))) v4i*)l, 0, 0);
}
__device__ __forceinline__ void wait_async() {
    asm volatile("s_wait_asynccnt 0x0" ::: "memory");
}
#else
#define HAVE_ASYNC 0
__device__ __forceinline__ void async_cp16(const _Float16* g, _Float16* l) {
    *(uint4*)l = *(const uint4*)g;
}
__device__ __forceinline__ void wait_async() {}
#endif

// ---------------------------------------------------------------------------
// WMMA wrapper: D = A(16x32 f16) * B(32x16 f16) + C(16x16 f32)
// ---------------------------------------------------------------------------
__device__ __forceinline__ v8f wmma_f16(v16h a, v16h b, v8f c) {
    return __builtin_amdgcn_wmma_f32_16x16x32_f16(
        false, a, false, b, (short)0, c, false, false);
}

// A-fragment (and N-major B-fragment) load from a 32-half-wide K chunk.
// ISA 7.12.2: lane<16 -> K in {0..7,16..23}; lane>=16 -> {8..15,24..31}
__device__ __forceinline__ v16h ld_frag(const _Float16* row_ptr, int chunk, int hsel) {
    const uint32_t* p = (const uint32_t*)row_ptr;
    FragH f;
    const int base = chunk * 16 + hsel * 4;   // dword offsets
#pragma unroll
    for (int j = 0; j < 4; ++j) {
        f.u[j]     = p[base + j];
        f.u[4 + j] = p[base + 8 + j];
    }
    return f.h;
}

// ---------------------------------------------------------------------------
// One-time precision/layout conversion kernels
// ---------------------------------------------------------------------------
__global__ __launch_bounds__(256)
void cvt_f16(const float* __restrict__ in, _Float16* __restrict__ out, int n) {
    int i = (blockIdx.x * 256 + threadIdx.x) * 4;
    if (i < n) {
        float4 f = *(const float4*)(in + i);
        PackH p0, p1;
        p0.h[0] = (_Float16)f.x; p0.h[1] = (_Float16)f.y;
        p1.h[0] = (_Float16)f.z; p1.h[1] = (_Float16)f.w;
        *(uint2*)(out + i) = make_uint2(p0.u, p1.u);
    }
}

// out[C][R] = (f16) in[R][C]   (LDS-tiled 32x32 transpose, coalesced both ways)
__global__ __launch_bounds__(256)
void cvt_T(const float* __restrict__ in, _Float16* __restrict__ out, int R, int C) {
    __shared__ _Float16 t[32][33];
    int br = blockIdx.y * 32, bc = blockIdx.x * 32;
    int lx = threadIdx.x & 31, ly = threadIdx.x >> 5;
#pragma unroll
    for (int i = 0; i < 32; i += 8)
        t[ly + i][lx] = (_Float16)in[(size_t)(br + ly + i) * C + bc + lx];
    __syncthreads();
#pragma unroll
    for (int i = 0; i < 32; i += 8)
        out[(size_t)(bc + ly + i) * R + br + lx] = t[lx][ly + i];
}

// ---------------------------------------------------------------------------
// WMMA GEMM: C[M,N] = A[M,K](f16) * Bt[N,K](f16, N-major), K step 32.
// Workgroup tile 128x256, 8 waves as 2x4, wave tile 64x64 (16 wmma / K-step).
// Double-buffered LDS staging (async-to-LDS when available).
//   EPI==0: dense f32 store to Cout
//   EPI==1: scatter f16 into q/k/v [B,H,T,HD]
// ---------------------------------------------------------------------------
template <int EPI>
__global__ __launch_bounds__(256)
void gemm_wmma(const _Float16* __restrict__ A, const _Float16* __restrict__ Bt,
               int M, int N, int K,
               float* __restrict__ Cout,
               _Float16* __restrict__ qo, _Float16* __restrict__ ko,
               _Float16* __restrict__ vo)
{
    __shared__ _Float16 Ah[2][128][32];
    __shared__ _Float16 Bh[2][256][32];

    const int tid  = threadIdx.x;
    const int bn   = blockIdx.x * 256;
    const int bm   = blockIdx.y * 128;
    const int wave = tid >> 5;
    const int lane = tid & 31;
    const int hsel = lane >> 4;
    const int lrow = lane & 15;
    const int wr   = wave >> 2;  // 0..1 : 64-row strip
    const int wc   = wave & 3;   // 0..3 : 64-col strip

    auto stage = [&](int buf, int kt) {
#pragma unroll
        for (int i = 0; i < 2; ++i) {              // A: 128 rows x 4 chunks
            int idx = tid + i * 256;
            int r = idx >> 2, c = idx & 3;
            async_cp16(A + (size_t)(bm + r) * K + kt + c * 8, &Ah[buf][r][c * 8]);
        }
#pragma unroll
        for (int i = 0; i < 4; ++i) {              // B: 256 rows x 4 chunks
            int idx = tid + i * 256;
            int r = idx >> 2, c = idx & 3;
            async_cp16(Bt + (size_t)(bn + r) * K + kt + c * 8, &Bh[buf][r][c * 8]);
        }
    };

    v8f acc[4][4] = {};

    stage(0, 0);
    wait_async();
    __syncthreads();

    int cur = 0;
    for (int kt = 0; kt < K; kt += 32) {
        if (kt + 32 < K) stage(cur ^ 1, kt + 32);   // overlap with compute

        v16h af[4], bf[4];
#pragma unroll
        for (int m = 0; m < 4; ++m)
            af[m] = ld_frag(&Ah[cur][wr * 64 + m * 16 + lrow][0], 0, hsel);
#pragma unroll
        for (int n = 0; n < 4; ++n)
            bf[n] = ld_frag(&Bh[cur][wc * 64 + n * 16 + lrow][0], 0, hsel);
#pragma unroll
        for (int m = 0; m < 4; ++m)
#pragma unroll
            for (int n = 0; n < 4; ++n)
                acc[m][n] = wmma_f16(af[m], bf[n], acc[m][n]);

        wait_async();
        __syncthreads();
        cur ^= 1;
    }

    // ---- epilogue ----
#pragma unroll
    for (int m = 0; m < 4; ++m) {
#pragma unroll
        for (int n = 0; n < 4; ++n) {
#pragma unroll
            for (int r = 0; r < 8; ++r) {
                int grow = bm + wr * 64 + m * 16 + r + (hsel ? 8 : 0);
                int gcol = bn + wc * 64 + n * 16 + lrow;
                float val = acc[m][n][r];
                if (EPI == 0) {
                    Cout[(size_t)grow * N + gcol] = val;
                } else {
                    int s   = gcol >> 10;      // (3, H, HD) decomposition
                    int rem = gcol & 1023;
                    int h   = rem >> 7;
                    int hd  = rem & 127;
                    int b   = grow >> 10;      // (B, T)
                    int t   = grow & 1023;
                    _Float16* dst = (s == 0) ? qo : (s == 1) ? ko : vo;
                    dst[(((b * H_) + h) * T_ + t) * HD_ + hd] = (_Float16)val;
                }
            }
        }
    }
}

// ---------------------------------------------------------------------------
// Flash attention with rank-1 trigram bias + causal mask.
// Block = 128 threads (4 waves); block owns (b,h) x 64 query rows.
// ---------------------------------------------------------------------------
__global__ __launch_bounds__(128)
void attn_kernel(const _Float16* __restrict__ Q, const _Float16* __restrict__ Kb,
                 const _Float16* __restrict__ Vb,
                 const float* __restrict__ head_scales,
                 const float* __restrict__ head_dirs,
                 _Float16* __restrict__ AO)   // [B,T,D]
{
    __shared__ _Float16 Ks[64][128];      // key-major (pure copy -> async)
    __shared__ _Float16 Vt[128][64];      // hd-major (transposed in regs)
    __shared__ _Float16 Ps[4][16][64];    // per-wave P tile
    __shared__ float    kproj[64];

    const int bh    = blockIdx.y;
    const int b     = bh >> 3;
    const int h     = bh & 7;
    const int qtile = blockIdx.x * 64;
    const int tid   = threadIdx.x;
    const int wave  = tid >> 5;
    const int lane  = tid & 31;
    const int hsel  = lane >> 4;
    const int lrow  = lane & 15;
    const int off   = hsel * 8;

    const _Float16* qg = Q  + (size_t)bh * T_ * HD_;
    const _Float16* kg = Kb + (size_t)bh * T_ * HD_;
    const _Float16* vg = Vb + (size_t)bh * T_ * HD_;

    const float hscale = head_scales[h];
    const float d0 = head_dirs[h * 3 + 0];
    const float d1 = head_dirs[h * 3 + 1];
    const float d2 = head_dirs[h * 3 + 2];
    const float sm_scale = 0.08838834764831845f;  // 1/sqrt(128)

    const int m_base = qtile + wave * 16;

    v16h qf[4];
    {
        const _Float16* qrow = qg + (size_t)(m_base + lrow) * HD_;
#pragma unroll
        for (int kc = 0; kc < 4; ++kc) qf[kc] = ld_frag(qrow, kc, hsel);
    }
    float qproj[8];
#pragma unroll
    for (int r = 0; r < 8; ++r) {
        const _Float16* p = qg + (size_t)(m_base + r + off) * HD_;
        qproj[r] = (float)p[0] * d0 + (float)p[1] * d1 + (float)p[2] * d2;
    }

    v8f   o[8] = {};
    float rowmax[8], rowsum[8];
#pragma unroll
    for (int r = 0; r < 8; ++r) { rowmax[r] = -__builtin_inff(); rowsum[r] = 0.f; }

    const int ntiles = qtile / 64 + 1;
    for (int ktile = 0; ktile < ntiles; ++ktile) {
        const int kbase = ktile * 64;

        // ---- stage K (async-capable pure copy) ----
#pragma unroll
        for (int i = 0; i < 8; ++i) {
            int idx = tid + i * 128;
            int r = idx >> 4, c = idx & 15;       // 64 rows x 16 b128 chunks
            async_cp16(kg + (size_t)(kbase + r) * HD_ + c * 8, &Ks[r][c * 8]);
        }
        // ---- stage V transposed (register path) ----
        const uint32_t* vgl = (const uint32_t*)(vg + (size_t)kbase * HD_);
#pragma unroll
        for (int i = 0; i < 32; ++i) {
            int idx = tid + i * 128;
            int r = idx >> 6, c = idx & 63;
            PackH pk; pk.u = vgl[r * 64 + c];
            Vt[2 * c + 0][r] = pk.h[0];
            Vt[2 * c + 1][r] = pk.h[1];
        }
        if (tid < 64) {
            const _Float16* p = kg + (size_t)(kbase + tid) * HD_;
            kproj[tid] = (float)p[0] * d0 + (float)p[1] * d1 + (float)p[2] * d2;
        }
        // prefetch next tile toward L2 while this one computes
        if (ktile + 1 < ntiles) {
            __builtin_prefetch(kg + (size_t)(kbase + 64) * HD_ + tid * 64, 0, 0);
            __builtin_prefetch(vg + (size_t)(kbase + 64) * HD_ + tid * 64, 0, 0);
        }
        wait_async();
        __syncthreads();

        // ---- S = Q K^T : preload 4 K-frags per N-subtile, then burst ----
        v8f sa[4] = {};
#pragma unroll
        for (int ns = 0; ns < 4; ++ns) {
            v16h kf[4];
#pragma unroll
            for (int kc = 0; kc < 4; ++kc)
                kf[kc] = ld_frag(&Ks[ns * 16 + lrow][0], kc, hsel);
#pragma unroll
            for (int kc = 0; kc < 4; ++kc)
                sa[ns] = wmma_f16(qf[kc], kf[kc], sa[ns]);
        }

        // ---- bias + causal mask + running max ----
        float pv[4][8], tmax[8];
#pragma unroll
        for (int r = 0; r < 8; ++r) tmax[r] = -__builtin_inff();
#pragma unroll
        for (int ns = 0; ns < 4; ++ns) {
            float kp    = kproj[ns * 16 + lrow];
            int   kgidx = kbase + ns * 16 + lrow;
#pragma unroll
            for (int r = 0; r < 8; ++r) {
                int   qgidx = m_base + r + off;
                float s = sa[ns][r] * sm_scale + hscale * qproj[r] * kp;
                s = (kgidx <= qgidx) ? s : -__builtin_inff();
                pv[ns][r] = s;
                tmax[r] = fmaxf(tmax[r], s);
            }
        }
#pragma unroll
        for (int r = 0; r < 8; ++r)
#pragma unroll
            for (int msk = 1; msk < 16; msk <<= 1)
                tmax[r] = fmaxf(tmax[r], __shfl_xor(tmax[r], msk, 16));

        float alpha[8], psum[8];
#pragma unroll
        for (int r = 0; r < 8; ++r) {
            float mnew = fmaxf(rowmax[r], tmax[r]);
            alpha[r] = __expf(rowmax[r] - mnew);
            rowmax[r] = mnew;
            psum[r] = 0.f;
        }
#pragma unroll
        for (int ns = 0; ns < 4; ++ns)
#pragma unroll
            for (int r = 0; r < 8; ++r) {
                float p = __expf(pv[ns][r] - rowmax[r]);
                pv[ns][r] = p;
                psum[r] += p;
            }
#pragma unroll
        for (int r = 0; r < 8; ++r)
#pragma unroll
            for (int msk = 1; msk < 16; msk <<= 1)
                psum[r] += __shfl_xor(psum[r], msk, 16);
#pragma unroll
        for (int r = 0; r < 8; ++r) rowsum[r] = rowsum[r] * alpha[r] + psum[r];
#pragma unroll
        for (int n = 0; n < 8; ++n)
#pragma unroll
            for (int r = 0; r < 8; ++r) o[n][r] *= alpha[r];

        // ---- P -> LDS (wave-private) -> A-fragments ----
#pragma unroll
        for (int ns = 0; ns < 4; ++ns)
#pragma unroll
            for (int r = 0; r < 8; ++r)
                Ps[wave][r + off][ns * 16 + lrow] = (_Float16)pv[ns][r];
        asm volatile("s_wait_dscnt 0" ::: "memory");  // wave-local LDS RAW fence

        v16h pf[2];
#pragma unroll
        for (int c = 0; c < 2; ++c)
            pf[c] = ld_frag(&Ps[wave][lrow][0], c, hsel);

        // ---- O += P V : preload both V-frags per HD-subtile ----
#pragma unroll
        for (int n = 0; n < 8; ++n) {
            v16h vf[2];
#pragma unroll
            for (int c = 0; c < 2; ++c)
                vf[c] = ld_frag(&Vt[n * 16 + lrow][0], c, hsel);
#pragma unroll
            for (int c = 0; c < 2; ++c)
                o[n] = wmma_f16(pf[c], vf[c], o[n]);
        }
        __syncthreads();
    }

    // ---- normalize and store f16 in [B,T,D] layout ----
#pragma unroll
    for (int r = 0; r < 8; ++r) {
        float inv = 1.f / rowsum[r];
        int   t   = m_base + r + off;
#pragma unroll
        for (int n = 0; n < 8; ++n) {
            int d = h * HD_ + n * 16 + lrow;
            AO[(size_t)(b * T_ + t) * D_ + d] = (_Float16)(o[n][r] * inv);
        }
    }
}

// ---------------------------------------------------------------------------
extern "C" void kernel_launch(void* const* d_in, const int* in_sizes, int n_in,
                              void* d_out, int out_size, void* d_ws, size_t ws_size,
                              hipStream_t stream)
{
    const float* x           = (const float*)d_in[0];   // [B,T,D]
    const float* Wqkv        = (const float*)d_in[1];   // [D,3D]
    const float* Wout        = (const float*)d_in[2];   // [D,D]
    const float* head_scales = (const float*)d_in[3];   // [H]
    const float* head_dirs   = (const float*)d_in[4];   // [H,3]

    const size_t per = (size_t)B_ * H_ * T_ * HD_;      // 4,194,304 halves
    _Float16* xh  = (_Float16*)d_ws;                    // [4096,1024]
    _Float16* wqT = xh  + (size_t)4096 * 1024;          // [3072,1024] (N-major)
    _Float16* woT = wqT + (size_t)3072 * 1024;          // [1024,1024] (N-major)
    _Float16* q   = woT + (size_t)1024 * 1024;
    _Float16* k   = q + per;
    _Float16* v   = k + per;
    _Float16* ao  = v + per;                            // [B,T,D] f16

    // 0) one-time precision/layout conversion
    cvt_f16<<<4096, 256, 0, stream>>>(x, xh, 4096 * 1024);
    cvt_T<<<dim3(3072 / 32, 1024 / 32), 256, 0, stream>>>(Wqkv, wqT, 1024, 3072);
    cvt_T<<<dim3(1024 / 32, 1024 / 32), 256, 0, stream>>>(Wout, woT, 1024, 1024);

    // 1) QKV GEMM -> scatter q/k/v f16 [B,H,T,HD]
    gemm_wmma<1><<<dim3(3072 / 256, 4096 / 128), 256, 0, stream>>>(
        xh, wqT, B_ * T_, 3 * D_, D_, nullptr, q, k, v);

    // 2) Flash attention with trigram bias -> ao f16 [B,T,D]
    attn_kernel<<<dim3(T_ / 64, B_ * H_), 128, 0, stream>>>(
        q, k, v, head_scales, head_dirs, ao);

    // 3) Out projection -> d_out f32
    gemm_wmma<0><<<dim3(1024 / 256, 4096 / 128), 256, 0, stream>>>(
        ao, woT, B_ * T_, D_, D_, (float*)d_out, nullptr, nullptr, nullptr);
}
